// BasicRNN_31207232373446
// MI455X (gfx1250) — matrix operand order repaired and checked by reading, not compiled, not run on hardware
//
#include <hip/hip_runtime.h>

// ---------------------------------------------------------------------------
// Types for CDNA5 WMMA (wave32): bf16 operands, fp32 accumulate.
// ---------------------------------------------------------------------------
typedef __bf16 bf16_t;
typedef __attribute__((ext_vector_type(16))) __bf16 v16bf;
typedef __attribute__((ext_vector_type(8)))  __bf16 v8bf;
typedef __attribute__((ext_vector_type(8)))  float  v8f;

#define VOCAB  32000
#define HIDDEN 1024
#define BATCH  16
#define SEQL   256
#define MROWS  (BATCH * SEQL)   // 4096

__device__ __forceinline__ v16bf cat8(v8bf lo, v8bf hi) {
    return __builtin_shufflevector(lo, hi, 0,1,2,3,4,5,6,7,8,9,10,11,12,13,14,15);
}

__device__ __forceinline__ v8f wmma_bf16(v16bf a, v16bf b, v8f c) {
    // D = A(16x32 bf16) * B(32x16 bf16) + C(16x16 f32)
    return __builtin_amdgcn_wmma_f32_16x16x32_bf16(
        /*neg_a=*/false, a, /*neg_b=*/false, b,
        /*c_mod=*/(short)0, c, /*reuse_a=*/false, /*reuse_b=*/false);
}

// ---------------------------------------------------------------------------
// CDNA5 async global->LDS copy (ASYNCcnt-tracked, bypasses VGPR staging).
// ---------------------------------------------------------------------------
typedef __attribute__((address_space(3))) bf16_t* lds_bf16_ptr;

__device__ __forceinline__ void async_copy_b128(const bf16_t* g, bf16_t* l) {
    lds_bf16_ptr lp = (lds_bf16_ptr)l;   // addrspacecast -> 32-bit LDS offset
    asm volatile("global_load_async_to_lds_b128 %0, %1, off"
                 :: "v"(lp), "v"(g) : "memory");
}

__device__ __forceinline__ void wait_async0() {
#if __has_builtin(__builtin_amdgcn_s_wait_asynccnt)
    __builtin_amdgcn_s_wait_asynccnt(0);
#else
    asm volatile("s_wait_asynccnt 0x0" ::: "memory");
#endif
}

// ---------------------------------------------------------------------------
// Prep kernels
// ---------------------------------------------------------------------------

// Gather embedding rows (fp32) into bf16 activation matrix Xe[4096][1024].
__global__ __launch_bounds__(256)
void k_gather_embed(const int* __restrict__ x, const float* __restrict__ emb,
                    bf16_t* __restrict__ Xe) {
    const int row = blockIdx.x;                 // b*SEQL + s
    const int id  = x[row];
    const float* src = emb + (size_t)id * HIDDEN;
    bf16_t*      dst = Xe  + (size_t)row * HIDDEN;
    const int c = threadIdx.x * 4;
    float4 v = *(const float4*)(src + c);
    dst[c + 0] = (bf16_t)v.x; dst[c + 1] = (bf16_t)v.y;
    dst[c + 2] = (bf16_t)v.z; dst[c + 3] = (bf16_t)v.w;
}

// Transpose fp32 W[K][N] -> bf16 WT[N][K]  (BT layout for WMMA B fragments).
__global__ __launch_bounds__(256)
void k_transpose_to_bf16(const float* __restrict__ W, bf16_t* __restrict__ WT,
                         int K, int N) {
    const size_t total  = (size_t)K * N;
    const size_t stride = (size_t)gridDim.x * blockDim.x;
    for (size_t idx = (size_t)blockIdx.x * blockDim.x + threadIdx.x;
         idx < total; idx += stride) {
        const size_t k = idx / (size_t)N;
        const size_t n = idx % (size_t)N;
        WT[n * (size_t)K + k] = (bf16_t)W[idx];
    }
}

// Plain fp32 -> bf16 convert (initial hidden state).
__global__ __launch_bounds__(256)
void k_convert_bf16(const float* __restrict__ src, bf16_t* __restrict__ dst, int n) {
    const int i = blockIdx.x * blockDim.x + threadIdx.x;
    if (i < n) dst[i] = (bf16_t)src[i];
}

// ---------------------------------------------------------------------------
// Tiled WMMA GEMM:  C[M][N] (f32) = A[M][K] (bf16) * BT[N][K]^T (bf16) + bias
// Block tile 128x128x32, 256 threads = 8 waves (wave32), wave grid 2x4,
// each wave computes a 64x32 patch = 4x2 v_wmma_f32_16x16x32_bf16 tiles.
// Double-buffered LDS filled by GLOBAL_LOAD_ASYNC_TO_LDS_B128 (ASYNCcnt).
// ---------------------------------------------------------------------------
__global__ __launch_bounds__(256)
void k_wmma_gemm_bt_bias(const bf16_t* __restrict__ A,
                         const bf16_t* __restrict__ BT,
                         const float*  __restrict__ bias,
                         float* __restrict__ C,
                         int M, int N, int K) {
    constexpr int BM = 128, BN = 128, BK = 32, PAD = 8, LDW = BK + PAD;
    __shared__ __align__(16) bf16_t As[2][BM][LDW];
    __shared__ __align__(16) bf16_t Bs[2][BN][LDW];

    const int tid   = threadIdx.x;
    const int wave  = tid >> 5;          // 0..7
    const int lane  = tid & 31;
    const int waveM = wave >> 2;         // 0..1 -> 64-row slab
    const int waveN = wave & 3;          // 0..3 -> 32-col slab
    const int half  = lane >> 4;         // hi/lo 16 lanes
    const int l16   = lane & 15;

    const int bm = blockIdx.y * BM;
    const int bn = blockIdx.x * BN;

    // Per-thread staging slice: 32 bytes of A tile + 32 bytes of BT tile.
    const int srow = tid >> 1;
    const int scol = (tid & 1) * 16;
    const bf16_t* gA = A  + (size_t)(bm + srow) * K + scol;
    const bf16_t* gB = BT + (size_t)(bn + srow) * K + scol;

    v8f acc[4][2];
#pragma unroll
    for (int i = 0; i < 4; ++i)
#pragma unroll
        for (int j = 0; j < 2; ++j)
            acc[i][j] = (v8f){0.f,0.f,0.f,0.f,0.f,0.f,0.f,0.f};

    // Prefetch k-tile 0 into buffer 0.
    async_copy_b128(gA,     &As[0][srow][scol]);
    async_copy_b128(gA + 8, &As[0][srow][scol + 8]);
    async_copy_b128(gB,     &Bs[0][srow][scol]);
    async_copy_b128(gB + 8, &Bs[0][srow][scol + 8]);
    wait_async0();
    __syncthreads();

    const int KT = K / BK;
    for (int kt = 0; kt < KT; ++kt) {
        const int cur = kt & 1;
        const int nxt = cur ^ 1;

        // Kick off async fill of the next k-tile while we compute this one.
        if (kt + 1 < KT) {
            const int k0 = (kt + 1) * BK;
            async_copy_b128(gA + k0,     &As[nxt][srow][scol]);
            async_copy_b128(gA + k0 + 8, &As[nxt][srow][scol + 8]);
            async_copy_b128(gB + k0,     &Bs[nxt][srow][scol]);
            async_copy_b128(gB + k0 + 8, &Bs[nxt][srow][scol + 8]);
        }

        // B fragments: lane holds column n, K chunks {half*8..+7} and {16+half*8..+7}.
        v16bf bfrag[2];
#pragma unroll
        for (int nt = 0; nt < 2; ++nt) {
            const int n = waveN * 32 + nt * 16 + l16;
            const bf16_t* p = &Bs[cur][n][half * 8];
            bfrag[nt] = cat8(*(const v8bf*)p, *(const v8bf*)(p + 16));
        }
#pragma unroll
        for (int mt = 0; mt < 4; ++mt) {
            // A fragment: lane row m, K chunks {half*8..+7} and {16+half*8..+7}.
            const int m = waveM * 64 + mt * 16 + l16;
            const bf16_t* p = &As[cur][m][0];
            v16bf afrag = cat8(*(const v8bf*)(p + half * 8),
                               *(const v8bf*)(p + 16 + half * 8));
#pragma unroll
            for (int nt = 0; nt < 2; ++nt)
                acc[mt][nt] = wmma_bf16(afrag, bfrag[nt], acc[mt][nt]);
        }

        if (kt + 1 < KT) {
            wait_async0();       // our async writes to the next buffer landed
            __syncthreads();     // every wave's writes landed + cur fully read
        }
    }

    // Epilogue: c[r] maps to tile row r + half*8; fuse bias add.
#pragma unroll
    for (int mt = 0; mt < 4; ++mt) {
#pragma unroll
        for (int nt = 0; nt < 2; ++nt) {
            const int col = bn + waveN * 32 + nt * 16 + l16;
            const float bv = bias[col];
#pragma unroll
            for (int r = 0; r < 8; ++r) {
                const int row = bm + waveM * 64 + mt * 16 + half * 8 + r;
                C[(size_t)row * N + col] = acc[mt][nt][r] + bv;
            }
        }
    }
}

// ---------------------------------------------------------------------------
// One recurrence step:  h_t = tanh(xw[:,t,:] + h_{t-1} @ W_hh)
// M = 16 (= batch) is exactly one WMMA tile; grid covers N=1024 columns.
// 16 blocks x 128 threads (4 waves); one 16x16 output tile per wave.
// ---------------------------------------------------------------------------
__global__ __launch_bounds__(128)
void k_rnn_step(const bf16_t* __restrict__ hprev, int hprev_stride,  // [16][H]
                const bf16_t* __restrict__ WhhT,                     // [H][H]
                const float*  __restrict__ xw,                       // [4096][H]
                bf16_t* __restrict__ hs,                             // [4096][H]
                float*  __restrict__ h_final,                        // [16][H]
                int t) {
    const int tid  = threadIdx.x;
    const int wave = tid >> 5;
    const int lane = tid & 31;
    const int half = lane >> 4;
    const int l16  = lane & 15;
    const int n0   = (blockIdx.x * 4 + wave) * 16;

    v8f acc = (v8f){0.f,0.f,0.f,0.f,0.f,0.f,0.f,0.f};
    for (int kk = 0; kk < HIDDEN / 32; ++kk) {
        const int k0 = kk * 32;
        const bf16_t* ap = hprev + (size_t)l16 * hprev_stride + k0;
        v16bf a = cat8(*(const v8bf*)(ap + half * 8),
                       *(const v8bf*)(ap + 16 + half * 8));
        const bf16_t* bp = WhhT + (size_t)(n0 + l16) * HIDDEN + k0 + half * 8;
        v16bf b = cat8(*(const v8bf*)(bp), *(const v8bf*)(bp + 16));
        acc = wmma_bf16(a, b, acc);
    }

#pragma unroll
    for (int r = 0; r < 8; ++r) {
        const int bidx = half * 8 + r;               // batch row
        const int n    = n0 + l16;
        const size_t row = (size_t)bidx * SEQL + t;  // [B,S] flattening
        const float v = tanhf(acc[r] + xw[row * HIDDEN + n]);
        hs[row * HIDDEN + n]               = (bf16_t)v;
        h_final[(size_t)bidx * HIDDEN + n] = v;       // last step (t=255) wins
    }
}

// ---------------------------------------------------------------------------
// Host-side orchestration
// ---------------------------------------------------------------------------
extern "C" void kernel_launch(void* const* d_in, const int* in_sizes, int n_in,
                              void* d_out, int out_size, void* d_ws, size_t ws_size,
                              hipStream_t stream) {
    (void)in_sizes; (void)n_in; (void)out_size; (void)ws_size;

    const int*   x    = (const int*)  d_in[0];   // [B,S]
    const float* h    = (const float*)d_in[1];   // [B,H]
    const float* emb  = (const float*)d_in[2];   // [V,H]
    const float* w_ih = (const float*)d_in[3];   // [H,H]
    const float* w_hh = (const float*)d_in[4];   // [H,H]
    const float* b_h  = (const float*)d_in[5];   // [H]
    const float* w_ho = (const float*)d_in[6];   // [H,V]
    const float* b_o  = (const float*)d_in[7];   // [V]

    float* out     = (float*)d_out;                         // [B,S,V]
    float* h_final = out + (size_t)MROWS * VOCAB;           // [B,H]

    // Workspace layout (all 16B aligned).
    char* ws = (char*)d_ws;
    size_t off = 0;
    bf16_t* Xe    = (bf16_t*)(ws + off); off += (size_t)MROWS  * HIDDEN * 2; //  8 MB
    bf16_t* WihT  = (bf16_t*)(ws + off); off += (size_t)HIDDEN * HIDDEN * 2; //  2 MB
    bf16_t* WhhT  = (bf16_t*)(ws + off); off += (size_t)HIDDEN * HIDDEN * 2; //  2 MB
    bf16_t* WhoT  = (bf16_t*)(ws + off); off += (size_t)VOCAB  * HIDDEN * 2; // 64 MB
    float*  xw    = (float*) (ws + off); off += (size_t)MROWS  * HIDDEN * 4; // 16 MB
    bf16_t* hs    = (bf16_t*)(ws + off); off += (size_t)MROWS  * HIDDEN * 2; //  8 MB
    bf16_t* h0    = (bf16_t*)(ws + off); off += (size_t)BATCH  * HIDDEN * 2; // 32 KB

    // --- Prep: gather + fp32->bf16 conversions / transposes -----------------
    k_gather_embed<<<MROWS, 256, 0, stream>>>(x, emb, Xe);
    k_transpose_to_bf16<<<2048, 256, 0, stream>>>(w_ih, WihT, HIDDEN, HIDDEN);
    k_transpose_to_bf16<<<2048, 256, 0, stream>>>(w_hh, WhhT, HIDDEN, HIDDEN);
    k_transpose_to_bf16<<<8192, 256, 0, stream>>>(w_ho, WhoT, HIDDEN, VOCAB);
    k_convert_bf16<<<(BATCH * HIDDEN + 255) / 256, 256, 0, stream>>>(h, h0, BATCH * HIDDEN);

    // --- Input projection: xw = Xe @ W_ih + b_h  (4096x1024x1024) ------------
    {
        dim3 grid(HIDDEN / 128, MROWS / 128);
        k_wmma_gemm_bt_bias<<<grid, 256, 0, stream>>>(Xe, WihT, b_h, xw,
                                                      MROWS, HIDDEN, HIDDEN);
    }

    // --- Sequential recurrence (256 dependent launches on the stream) --------
    for (int t = 0; t < SEQL; ++t) {
        const bf16_t* hprev  = (t == 0) ? h0 : (hs + (size_t)(t - 1) * HIDDEN);
        const int     stride = (t == 0) ? HIDDEN : SEQL * HIDDEN;
        k_rnn_step<<<HIDDEN / 64, 128, 0, stream>>>(hprev, stride, WhhT, xw,
                                                    hs, h_final, t);
    }

    // --- Output projection: out = hs @ W_ho + b_o  (4096x32000x1024) ---------
    {
        dim3 grid(VOCAB / 128, MROWS / 128);
        k_wmma_gemm_bt_bias<<<grid, 256, 0, stream>>>(hs, WhoT, b_o, out,
                                                      MROWS, VOCAB, HIDDEN);
    }
}
